// SlotAttention_75651553952006
// MI455X (gfx1250) — matrix operand order repaired and checked by reading, not compiled, not run on hardware
//
#include <hip/hip_runtime.h>
#include <hip/hip_bf16.h>
#include <math.h>

// ---------------------------------------------------------------------------
// CDNA5 / gfx1250 slot-attention block, bf16 WMMA (v_wmma_f32_16x16x32_bf16)
// B=1024, S=16 slots, N=36 tokens, D=H=768, ITERS=3
// ---------------------------------------------------------------------------

typedef __bf16 bf16_t;
typedef __attribute__((ext_vector_type(16))) __bf16 v16bf;
typedef __attribute__((ext_vector_type(8)))  __bf16 v8bf;
typedef __attribute__((ext_vector_type(8)))  float  v8f;

#define BB 1024
#define SS 16
#define NN 36
#define DDIM 768
#define ITERS 3

// ---------------- WMMA fragment helpers (contiguous k = 16*half + e in A and B) -----------

__device__ __forceinline__ v8f zero_v8f() {
    v8f z;
#pragma unroll
    for (int i = 0; i < 8; ++i) z[i] = 0.0f;
    return z;
}

__device__ __forceinline__ v16bf zero_v16bf() {
    v16bf z;
#pragma unroll
    for (int i = 0; i < 16; ++i) z[i] = (bf16_t)0.0f;
    return z;
}

// A matrix 16x32 (MxK), row-major with leading dim lda.
// lane L: m = L%16 ; half = L/16 ; elems e=0..15 -> K = k0 + half*16 + e  (contiguous)
__device__ __forceinline__ v16bf load_a_frag(const bf16_t* A, int lda,
                                             int m0, int k0, int lane) {
    int m = m0 + (lane & 15);
    int half = lane >> 4;
    const bf16_t* p = A + (size_t)m * lda + k0 + half * 16;
    v8bf lo = *(const v8bf*)p;
    v8bf hi = *(const v8bf*)(p + 8);
    v16bf r;
#pragma unroll
    for (int e = 0; e < 8; ++e) { r[e] = lo[e]; r[e + 8] = hi[e]; }
    return r;
}

// B matrix 32x16 (KxN), supplied in transposed storage W[n][k] (row-major, ld=ldk).
// lane L: n = n0 + L%16 ; half = L/16 ; elem e -> K = k0 + half*16 + e  (contiguous)
__device__ __forceinline__ v16bf load_bT_frag(const bf16_t* W, int ldk,
                                              int n0, int k0, int lane) {
    int n = n0 + (lane & 15);
    int half = lane >> 4;
    return *(const v16bf*)(W + (size_t)n * ldk + k0 + half * 16);
}

__device__ __forceinline__ v16bf load_bT_frag_guard(const bf16_t* W, int ldk,
                                                    int n0, int k0, int nmax,
                                                    int lane) {
    int n = n0 + (lane & 15);
    int half = lane >> 4;
    if (n < nmax)
        return *(const v16bf*)(W + (size_t)n * ldk + k0 + half * 16);
    return zero_v16bf();
}

// B matrix from k-major storage Bm[k][n] (row-major, ld=ldn), rows k >= kmax are zero.
__device__ __forceinline__ v16bf load_b_kmajor_guard(const bf16_t* Bm, int ldn,
                                                     int k0, int n0, int kmax,
                                                     int lane) {
    int n = n0 + (lane & 15);
    int half = lane >> 4;
    v16bf r;
#pragma unroll
    for (int e = 0; e < 16; ++e) {
        int k = k0 + half * 16 + e;
        r[e] = (k < kmax) ? Bm[(size_t)k * ldn + n] : (bf16_t)0.0f;
    }
    return r;
}

__device__ __forceinline__ v8f wmma_bf16(v16bf a, v16bf b, v8f c) {
    return __builtin_amdgcn_wmma_f32_16x16x32_bf16(
        /*neg_a=*/false, a, /*neg_b=*/false, b,
        /*c_mod=*/(short)0, c, /*reuse_a=*/false, /*reuse_b=*/false);
}

// ---------------- elementwise convert ---------------------------------------

__global__ void cvt_f32_to_bf16_kernel(const float* __restrict__ src,
                                       bf16_t* __restrict__ dst, int n) {
    int i = blockIdx.x * blockDim.x + threadIdx.x;
    if (i < n) dst[i] = (bf16_t)src[i];
}

// ---------------- LayerNorm (row = 768), one wave per row, bf16 out ---------

__global__ void layernorm_bf16_kernel(const float* __restrict__ x,
                                      const float* __restrict__ gamma,
                                      const float* __restrict__ beta,
                                      bf16_t* __restrict__ out, int rows) {
    int wave = threadIdx.x >> 5;
    int lane = threadIdx.x & 31;
    int row = blockIdx.x * (blockDim.x >> 5) + wave;
    if (row >= rows) return;
    const float* xr = x + (size_t)row * DDIM;
    float vals[24];
    float s = 0.0f;
#pragma unroll
    for (int i = 0; i < 24; ++i) {
        float v = xr[lane + i * 32];
        vals[i] = v;
        s += v;
    }
#pragma unroll
    for (int off = 16; off > 0; off >>= 1) s += __shfl_xor(s, off, 32);
    float mu = s * (1.0f / DDIM);
    float s2 = 0.0f;
#pragma unroll
    for (int i = 0; i < 24; ++i) {
        float d = vals[i] - mu;
        s2 += d * d;
    }
#pragma unroll
    for (int off = 16; off > 0; off >>= 1) s2 += __shfl_xor(s2, off, 32);
    float rstd = rsqrtf(s2 * (1.0f / DDIM) + 1e-5f);
    bf16_t* orow = out + (size_t)row * DDIM;
#pragma unroll
    for (int i = 0; i < 24; ++i) {
        int c = lane + i * 32;
        orow[c] = (bf16_t)((vals[i] - mu) * rstd * gamma[c] + beta[c]);
    }
}

// ---------------- generic WMMA GEMM: C = act(A @ W^T + bias) [+resid] -------
// A: [M,K] bf16 row-major, W: [N,K] bf16 row-major (torch linear weight).
// Each wave -> one 16x64 tile; block = 4 waves = 64x64 tile.
// EPI: 0 = fp32 store, 1 = bf16 store, 2 = relu+bf16, 3 = fp32 + resid.

template <int EPI>
__global__ void gemm_wmma_kernel(const bf16_t* __restrict__ A,
                                 const bf16_t* __restrict__ W,
                                 const float* __restrict__ bias,
                                 float* __restrict__ outF,
                                 bf16_t* __restrict__ outB,
                                 const float* __restrict__ resid,
                                 int M, int N, int K) {
    int lane = threadIdx.x & 31;
    int wave = threadIdx.x >> 5;
    int m0 = (blockIdx.y * 4 + wave) * 16;
    int n0 = blockIdx.x * 64;
    if (m0 >= M) return;

    v8f acc[4];
#pragma unroll
    for (int t = 0; t < 4; ++t) acc[t] = zero_v8f();

    for (int k0 = 0; k0 < K; k0 += 32) {
        v16bf a = load_a_frag(A, K, m0, k0, lane);
#pragma unroll
        for (int t = 0; t < 4; ++t) {
            v16bf b = load_bT_frag(W, K, n0 + t * 16, k0, lane);
            acc[t] = wmma_bf16(a, b, acc[t]);
        }
    }

    int half = lane >> 4, nc = lane & 15;
#pragma unroll
    for (int t = 0; t < 4; ++t) {
        int n = n0 + t * 16 + nc;
        float bv = bias[n];
#pragma unroll
        for (int v = 0; v < 8; ++v) {
            int m = m0 + v + half * 8;
            size_t idx = (size_t)m * N + n;
            float val = acc[t][v] + bv;
            if (EPI == 0) outF[idx] = val;
            else if (EPI == 1) outB[idx] = (bf16_t)val;
            else if (EPI == 2) outB[idx] = (bf16_t)fmaxf(val, 0.0f);
            else outF[idx] = val + resid[idx];
        }
    }
}

// ---------------- per-batch attention core ----------------------------------
// dots = Q K^T * scale ; mask rows -> -inf ; softmax over SLOT axis (per column)
// updates = attn @ V. One workgroup (128 thr) per batch. N padded 36->48/64.

__global__ void slot_attn_kernel(const bf16_t* __restrict__ Q,
                                 const bf16_t* __restrict__ Kb_,
                                 const bf16_t* __restrict__ Vb_,
                                 const unsigned char* __restrict__ mask,
                                 float* __restrict__ attn_out,
                                 bf16_t* __restrict__ upd_out,
                                 float scale) {
    __shared__ __align__(64) float  dots[SS][48];
    __shared__ __align__(64) bf16_t attn_sh[SS][64];

    int b = blockIdx.x;
    int lane = threadIdx.x & 31;
    int wave = threadIdx.x >> 5;
    const bf16_t* Qb = Q  + (size_t)b * SS * DDIM;
    const bf16_t* Kk = Kb_ + (size_t)b * NN * DDIM;
    const bf16_t* Vv = Vb_ + (size_t)b * NN * DDIM;

    // Phase 1: dots tiles (waves 0..2, each a 16x16 j-tile; j padded to 48)
    if (wave < 3) {
        v8f acc = zero_v8f();
        int n0 = wave * 16;
        for (int k0 = 0; k0 < DDIM; k0 += 32) {
            v16bf a = load_a_frag(Qb, DDIM, 0, k0, lane);
            v16bf bb = load_bT_frag_guard(Kk, DDIM, n0, k0, NN, lane);
            acc = wmma_bf16(a, bb, acc);
        }
        int half = lane >> 4, j = n0 + (lane & 15);
#pragma unroll
        for (int v = 0; v < 8; ++v) dots[v + half * 8][j] = acc[v] * scale;
    }
    __syncthreads();

    // Phase 2: softmax over slot axis per column j (threads 0..35), threads 36..63 zero pad.
    int t = threadIdx.x;
    if (t < 64) {
        if (t < NN) {
            float col[SS];
            float mx = -__builtin_inff();
#pragma unroll
            for (int i = 0; i < SS; ++i) {
                float dv = mask[b * SS + i] ? -__builtin_inff() : dots[i][t];
                col[i] = dv;
                mx = fmaxf(mx, dv);
            }
            float sum = 0.0f;
#pragma unroll
            for (int i = 0; i < SS; ++i) {
                float e = expf(col[i] - mx);
                col[i] = e;
                sum += e;
            }
            float inv = 1.0f / sum;
#pragma unroll
            for (int i = 0; i < SS; ++i) {
                float a = col[i] * inv;
                attn_out[((size_t)b * SS + i) * NN + t] = a;
                attn_sh[i][t] = (bf16_t)a;
            }
        } else {
#pragma unroll
            for (int i = 0; i < SS; ++i) attn_sh[i][t] = (bf16_t)0.0f;
        }
    }
    __syncthreads();

    // Phase 3: updates = attn[16x64pad] @ V[36x768]; 4 waves x 3 chunks of 64.
    for (int c = 0; c < 3; ++c) {
        int n0 = (wave * 3 + c) * 64;
        v8f acc[4];
#pragma unroll
        for (int t4 = 0; t4 < 4; ++t4) acc[t4] = zero_v8f();
        for (int k0 = 0; k0 < 64; k0 += 32) {
            v16bf a = load_a_frag((const bf16_t*)attn_sh, 64, 0, k0, lane);
#pragma unroll
            for (int t4 = 0; t4 < 4; ++t4) {
                v16bf bb = load_b_kmajor_guard(Vv, DDIM, k0, n0 + t4 * 16, NN, lane);
                acc[t4] = wmma_bf16(a, bb, acc[t4]);
            }
        }
        int half = lane >> 4, nc = lane & 15;
#pragma unroll
        for (int t4 = 0; t4 < 4; ++t4) {
            int n = n0 + t4 * 16 + nc;
#pragma unroll
            for (int v = 0; v < 8; ++v) {
                int m = v + half * 8;
                upd_out[((size_t)b * SS + m) * DDIM + n] = (bf16_t)acc[t4][v];
            }
        }
    }
}

// ---------------- fused GRU cell --------------------------------------------
// Per wave: 16(m) x 16(d) output tile; 6 WMMA accumulators (i_r,i_z,i_n from
// updates@w_ih^T ; h_r,h_z,h_n from prev@w_hh^T). Gates applied in-register.
// Avoids materializing gi/gh (saves ~300 MB of HBM round trip).

__global__ void gru_fused_kernel(const bf16_t* __restrict__ upd,
                                 const bf16_t* __restrict__ prevb,
                                 const bf16_t* __restrict__ w_ih,
                                 const bf16_t* __restrict__ w_hh,
                                 const float* __restrict__ b_ih,
                                 const float* __restrict__ b_hh,
                                 const float* __restrict__ prevf,
                                 float* __restrict__ g_out,
                                 int M) {
    int lane = threadIdx.x & 31;
    int wave = threadIdx.x >> 5;
    int m0 = (blockIdx.y * 4 + wave) * 16;
    int d0 = blockIdx.x * 16;
    if (m0 >= M) return;

    v8f aIr = zero_v8f(), aIz = zero_v8f(), aIn = zero_v8f();
    v8f aHr = zero_v8f(), aHz = zero_v8f(), aHn = zero_v8f();

    for (int k0 = 0; k0 < DDIM; k0 += 32) {
        v16bf au = load_a_frag(upd,   DDIM, m0, k0, lane);
        v16bf ap = load_a_frag(prevb, DDIM, m0, k0, lane);
        v16bf b0 = load_bT_frag(w_ih, DDIM, d0,            k0, lane);
        aIr = wmma_bf16(au, b0, aIr);
        v16bf b1 = load_bT_frag(w_ih, DDIM, DDIM + d0,     k0, lane);
        aIz = wmma_bf16(au, b1, aIz);
        v16bf b2 = load_bT_frag(w_ih, DDIM, 2 * DDIM + d0, k0, lane);
        aIn = wmma_bf16(au, b2, aIn);
        v16bf b3 = load_bT_frag(w_hh, DDIM, d0,            k0, lane);
        aHr = wmma_bf16(ap, b3, aHr);
        v16bf b4 = load_bT_frag(w_hh, DDIM, DDIM + d0,     k0, lane);
        aHz = wmma_bf16(ap, b4, aHz);
        v16bf b5 = load_bT_frag(w_hh, DDIM, 2 * DDIM + d0, k0, lane);
        aHn = wmma_bf16(ap, b5, aHn);
    }

    int half = lane >> 4, d = d0 + (lane & 15);
    float bir = b_ih[d], biz = b_ih[DDIM + d], bin_ = b_ih[2 * DDIM + d];
    float bhr = b_hh[d], bhz = b_hh[DDIM + d], bhn = b_hh[2 * DDIM + d];
#pragma unroll
    for (int v = 0; v < 8; ++v) {
        int m = m0 + v + half * 8;
        float r = 1.0f / (1.0f + expf(-(aIr[v] + bir + aHr[v] + bhr)));
        float z = 1.0f / (1.0f + expf(-(aIz[v] + biz + aHz[v] + bhz)));
        float n = tanhf(aIn[v] + bin_ + r * (aHn[v] + bhn));
        float pv = prevf[(size_t)m * DDIM + d];
        g_out[(size_t)m * DDIM + d] = (1.0f - z) * n + z * pv;
    }
}

// ---------------------------------------------------------------------------

extern "C" void kernel_launch(void* const* d_in, const int* in_sizes, int n_in,
                              void* d_out, int out_size, void* d_ws, size_t ws_size,
                              hipStream_t stream) {
    (void)in_sizes; (void)n_in; (void)out_size; (void)ws_size;

    const float* cand_feat = (const float*)d_in[0];
    const float* pano_feat = (const float*)d_in[1];
    const unsigned char* cand_mask = (const unsigned char*)d_in[2];
    const float* wq = (const float*)d_in[3];  const float* bq = (const float*)d_in[4];
    const float* wk = (const float*)d_in[5];  const float* bk = (const float*)d_in[6];
    const float* wv = (const float*)d_in[7];  const float* bv = (const float*)d_in[8];
    const float* w_ih = (const float*)d_in[9];  const float* b_ih = (const float*)d_in[10];
    const float* w_hh = (const float*)d_in[11]; const float* b_hh = (const float*)d_in[12];
    const float* w1 = (const float*)d_in[13]; const float* b1 = (const float*)d_in[14];
    const float* w2 = (const float*)d_in[15]; const float* b2 = (const float*)d_in[16];
    const float* g_slots = (const float*)d_in[17]; const float* be_slots = (const float*)d_in[18];
    const float* g_ff = (const float*)d_in[19];    const float* be_ff = (const float*)d_in[20];
    const float* g_in = (const float*)d_in[21];    const float* be_in = (const float*)d_in[22];

    float* out = (float*)d_out;
    float* slots_f32 = out;                              // [B,S,D] working + final
    float* attn_out = out + (size_t)BB * SS * DDIM;      // [B,S,N]

    const int DxD = DDIM * DDIM;          // 589824
    const int D3xD = 3 * DDIM * DDIM;     // 1769472
    const size_t nBND = (size_t)BB * NN * DDIM;   // 28,311,552
    const size_t nBSD = (size_t)BB * SS * DDIM;   // 12,582,912

    char* wsp = (char*)d_ws;
    size_t off = 0;
    auto carve = [&](size_t bytes) -> void* {
        void* p = wsp + off;
        off = (off + bytes + 255) & ~(size_t)255;
        return p;
    };

    bf16_t* wq_b  = (bf16_t*)carve((size_t)DxD * 2);
    bf16_t* wk_b  = (bf16_t*)carve((size_t)DxD * 2);
    bf16_t* wv_b  = (bf16_t*)carve((size_t)DxD * 2);
    bf16_t* wih_b = (bf16_t*)carve((size_t)D3xD * 2);
    bf16_t* whh_b = (bf16_t*)carve((size_t)D3xD * 2);
    bf16_t* w1_b  = (bf16_t*)carve((size_t)DxD * 2);
    bf16_t* w2_b  = (bf16_t*)carve((size_t)DxD * 2);
    bf16_t* pano_ln = (bf16_t*)carve(nBND * 2);
    bf16_t* Kbuf    = (bf16_t*)carve(nBND * 2);
    bf16_t* Vbuf    = (bf16_t*)carve(nBND * 2);
    bf16_t* xln     = (bf16_t*)carve(nBSD * 2);
    bf16_t* Qbuf    = (bf16_t*)carve(nBSD * 2);
    bf16_t* updbuf  = (bf16_t*)carve(nBSD * 2);
    bf16_t* slots_b = (bf16_t*)carve(nBSD * 2);
    bf16_t* h1buf   = (bf16_t*)carve(nBSD * 2);
    float*  gbuf    = (float*) carve(nBSD * 4);

    auto cvt = [&](const float* s, bf16_t* d, int n) {
        cvt_f32_to_bf16_kernel<<<(n + 255) / 256, 256, 0, stream>>>(s, d, n);
    };

    // --- one-time: weights -> bf16, slots init, pano LN, K/V projections ---
    cvt(wq, wq_b, DxD);   cvt(wk, wk_b, DxD);   cvt(wv, wv_b, DxD);
    cvt(w_ih, wih_b, D3xD); cvt(w_hh, whh_b, D3xD);
    cvt(w1, w1_b, DxD);   cvt(w2, w2_b, DxD);

    hipMemcpyAsync(slots_f32, cand_feat, nBSD * sizeof(float),
                   hipMemcpyDeviceToDevice, stream);

    const int rowsBN = BB * NN;   // 36864
    const int rowsBS = BB * SS;   // 16384
    layernorm_bf16_kernel<<<rowsBN / 8, 256, 0, stream>>>(pano_feat, g_in, be_in,
                                                          pano_ln, rowsBN);

    dim3 gemmBlk(128);
    dim3 gridKV(DDIM / 64, rowsBN / 64);
    gemm_wmma_kernel<1><<<gridKV, gemmBlk, 0, stream>>>(
        pano_ln, wk_b, bk, nullptr, Kbuf, nullptr, rowsBN, DDIM, DDIM);
    gemm_wmma_kernel<1><<<gridKV, gemmBlk, 0, stream>>>(
        pano_ln, wv_b, bv, nullptr, Vbuf, nullptr, rowsBN, DDIM, DDIM);

    const float scale = 1.0f / sqrtf((float)DDIM);
    dim3 gridS(DDIM / 64, rowsBS / 64);

    for (int it = 0; it < ITERS; ++it) {
        // prev slots in bf16 for the GRU hidden GEMM
        cvt_f32_to_bf16_kernel<<<((int)nBSD + 255) / 256, 256, 0, stream>>>(
            slots_f32, slots_b, (int)nBSD);

        // q = LN(slots; g_slots) @ wq^T + bq
        layernorm_bf16_kernel<<<rowsBS / 8, 256, 0, stream>>>(
            slots_f32, g_slots, be_slots, xln, rowsBS);
        gemm_wmma_kernel<1><<<gridS, gemmBlk, 0, stream>>>(
            xln, wq_b, bq, nullptr, Qbuf, nullptr, rowsBS, DDIM, DDIM);

        // dots -> masked softmax over slots -> attn, updates = attn @ V
        slot_attn_kernel<<<BB, 128, 0, stream>>>(Qbuf, Kbuf, Vbuf, cand_mask,
                                                 attn_out, updbuf, scale);

        // fused GRUCell: g = (1-z)*n + z*prev
        gru_fused_kernel<<<dim3(DDIM / 16, rowsBS / 64), gemmBlk, 0, stream>>>(
            updbuf, slots_b, wih_b, whh_b, b_ih, b_hh, slots_f32, gbuf, rowsBS);

        // ff: slots = g + relu(LN(g; g_ff) @ w1^T + b1) @ w2^T + b2
        layernorm_bf16_kernel<<<rowsBS / 8, 256, 0, stream>>>(
            gbuf, g_ff, be_ff, xln, rowsBS);
        gemm_wmma_kernel<2><<<gridS, gemmBlk, 0, stream>>>(
            xln, w1_b, b1, nullptr, h1buf, nullptr, rowsBS, DDIM, DDIM);
        gemm_wmma_kernel<3><<<gridS, gemmBlk, 0, stream>>>(
            h1buf, w2_b, b2, slots_f32, nullptr, gbuf, rowsBS, DDIM, DDIM);
    }
}
